// GCNConvNet_34892314312705
// MI455X (gfx1250) — matrix operand order repaired and checked by reading, not compiled, not run on hardware
//
#include <hip/hip_runtime.h>
#include <cstdint>

// ---------------------------------------------------------------------------
// CDNA5 (gfx1250): bf16 WMMA GEMMs (TDM-staged LDS for the big node GEMMs),
// atomic graph aggregation, fused conv+pool tower, WMMA dense heads.
// ---------------------------------------------------------------------------

typedef __attribute__((ext_vector_type(16))) __bf16 v16bf;
typedef __attribute__((ext_vector_type(8)))  float  v8f;
typedef __attribute__((ext_vector_type(4)))  unsigned int u32x4;
typedef __attribute__((ext_vector_type(8)))  int i32x8;
typedef __attribute__((ext_vector_type(4)))  int i32x4;

union FragBF { v16bf v; uint32_t u[8]; };

__device__ __forceinline__ uint16_t f2bf(float f) {
  uint32_t u = __float_as_uint(f);
  uint32_t r = u + 0x7FFFu + ((u >> 16) & 1u);   // round-to-nearest-even
  return (uint16_t)(r >> 16);
}

// float (rows x cols) -> bf16 (rows x cols_pad), zero padded columns
__global__ void k_cvt_pad(const float* __restrict__ in, uint16_t* __restrict__ out,
                          int rows, int cols, int cols_pad) {
  long long t = (long long)blockIdx.x * blockDim.x + threadIdx.x;
  long long total = (long long)rows * cols_pad;
  if (t >= total) return;
  int c = (int)(t % cols_pad);
  long long r = t / cols_pad;
  out[t] = (c < cols) ? f2bf(in[r * (long long)cols + c]) : (uint16_t)0;
}

// weight (K x N) f32 row-major -> transposed padded bf16 (Np x Kp)
__global__ void k_cvt_wt(const float* __restrict__ w, uint16_t* __restrict__ out,
                         int K, int N, int Kp, int Np) {
  long long t = (long long)blockIdx.x * blockDim.x + threadIdx.x;
  long long total = (long long)Np * Kp;
  if (t >= total) return;
  int k = (int)(t % Kp);
  int n = (int)(t / Kp);
  out[t] = (k < K && n < N) ? f2bf(w[(long long)k * N + n]) : (uint16_t)0;
}

// ---------------- direct-global GEMM (small M) -----------------------------
// C (M x Np, f32) = A (M x Kp bf16 row-major) @ Bt^T (Bt: Np x Kp bf16)
__global__ __launch_bounds__(256) void k_gemm_bf16(
    const uint16_t* __restrict__ A, const uint16_t* __restrict__ Bt,
    float* __restrict__ C, int tiles_m, int tiles_n, int Kp, int Np) {
  int wid = (int)(((long long)blockIdx.x * blockDim.x + threadIdx.x) >> 5);
  if (wid >= tiles_m * tiles_n) return;          // wave-uniform: EXEC stays all-1s
  int tm = wid % tiles_m;
  int tn = wid / tiles_m;
  int lane = threadIdx.x & 31;
  int l15  = lane & 15;
  int lh   = (lane >> 4) & 1;
  const uint16_t* Ar = A  + (size_t)(tm * 16 + l15) * Kp;
  const uint16_t* Br = Bt + (size_t)(tn * 16 + l15) * Kp;
  v8f acc = {0.f, 0.f, 0.f, 0.f, 0.f, 0.f, 0.f, 0.f};
  for (int k0 = 0; k0 < Kp; k0 += 32) {
    FragBF a, b;
#pragma unroll
    for (int j = 0; j < 8; ++j) {
      int ka = k0 + ((j < 4) ? 0 : 16) + lh * 8 + (j & 3) * 2;   // A frag (ISA 7.12.2)
      int kb = k0 + ((j < 4) ? 0 : 16) + lh * 8 + (j & 3) * 2;   // B frag (same k order as A)
      a.u[j] = *(const uint32_t*)(Ar + ka);
      b.u[j] = *(const uint32_t*)(Br + kb);
    }
    acc = __builtin_amdgcn_wmma_f32_16x16x32_bf16(false, a.v, false, b.v,
                                                  (short)0, acc, false, false);
  }
  size_t cbase = (size_t)(tm * 16) * Np + (size_t)tn * 16 + l15;
#pragma unroll
  for (int j = 0; j < 8; ++j) {
    int mm = j + lh * 8;
    C[cbase + (size_t)mm * Np] = acc[j];
  }
}

// ---------------- TDM-staged LDS GEMM (M multiple of 80, Kp <= 320) --------
// Block: 256 thr (8 waves). Block tile: 80 M-rows x 8 N-tiles.
// One TDM descriptor loads the 80 x Kp bf16 A-slab into LDS; waves read A
// fragments from LDS (ds_load) and B fragments from global (weight, cached).
#define TDM_BM 80
__global__ __launch_bounds__(256) void k_gemm_tdm(
    const uint16_t* __restrict__ A, const uint16_t* __restrict__ Bt,
    float* __restrict__ C, int Kp, int Np, int tiles_n) {
  __shared__ uint16_t sA[TDM_BM * 320];          // 50 KB max
  int row0 = blockIdx.x * TDM_BM;
  int wave = threadIdx.x >> 5;
  int lane = threadIdx.x & 31;
  int l15  = lane & 15;
  int lh   = (lane >> 4) & 1;

  if (wave == 0) {
    // Build D# (ISA 8.3/8.4): 2-D tile, data_size=2B, tile = Kp x TDM_BM rows.
    unsigned long long ga = (unsigned long long)(const void*)(A + (size_t)row0 * Kp);
    unsigned lds = (unsigned)(unsigned long long)(uintptr_t)&sA[0]; // flat->LDS: addr[31:0]
    unsigned dim0 = (unsigned)Kp;
    unsigned dim1 = (unsigned)TDM_BM;
    unsigned long long st0 = (unsigned long long)Kp;
    u32x4 g0;
    g0.x = 1u;                                              // count=1 (valid user D#)
    g0.y = lds;                                             // lds_addr
    g0.z = (unsigned)(ga & 0xFFFFFFFFu);                    // global_addr[31:0]
    g0.w = (unsigned)((ga >> 32) & 0x01FFFFFFu) | (2u << 30); // addr[56:32] | type=2
    i32x8 g1;
    g1.s0 = (int)(1u << 16);                                // data_size=1 (2 bytes)
    g1.s1 = (int)((dim0 & 0xFFFFu) << 16);                  // tensor_dim0[15:0]
    g1.s2 = (int)((dim0 >> 16) | ((dim1 & 0xFFFFu) << 16)); // dim0[31:16] | dim1[15:0]
    g1.s3 = (int)((dim1 >> 16) | (dim0 << 16));             // dim1[31:16] | tile_dim0=Kp
    g1.s4 = (int)(dim1 & 0xFFFFu);                          // tile_dim1=BM, tile_dim2=0
    g1.s5 = (int)(st0 & 0xFFFFFFFFu);                       // tensor_dim0_stride lo
    g1.s6 = (int)((st0 >> 32) & 0xFFFFu);                   // stride hi | dim1_stride lo=0
    g1.s7 = 0;
    i32x4 z4 = {0, 0, 0, 0};
#if defined(__clang_major__) && (__clang_major__ >= 23)
    i32x8 z8 = {0, 0, 0, 0, 0, 0, 0, 0};
    __builtin_amdgcn_tensor_load_to_lds(g0, g1, z4, z4, z8, 0);
#else
    __builtin_amdgcn_tensor_load_to_lds(g0, g1, z4, z4, 0);
#endif
    __builtin_amdgcn_s_wait_tensorcnt(0);
  }
  __syncthreads();

  int tn  = blockIdx.y * 8 + wave;
  int tnc = (tn < tiles_n) ? tn : (tiles_n - 1);            // clamp (loads stay in-bounds)
  const uint16_t* Br = Bt + (size_t)(tnc * 16 + l15) * Kp;

  v8f acc[5];
#pragma unroll
  for (int ms = 0; ms < 5; ++ms) acc[ms] = (v8f){0.f,0.f,0.f,0.f,0.f,0.f,0.f,0.f};

  for (int k0 = 0; k0 < Kp; k0 += 32) {
    FragBF b;
#pragma unroll
    for (int j = 0; j < 8; ++j) {
      int kb = k0 + ((j < 4) ? 0 : 16) + lh * 8 + (j & 3) * 2;
      b.u[j] = *(const uint32_t*)(Br + kb);
    }
#pragma unroll
    for (int ms = 0; ms < 5; ++ms) {
      FragBF a;
      const uint16_t* Ar = &sA[(ms * 16 + l15) * Kp];
#pragma unroll
      for (int j = 0; j < 8; ++j) {
        int ka = k0 + ((j < 4) ? 0 : 16) + lh * 8 + (j & 3) * 2;
        a.u[j] = *(const uint32_t*)(Ar + ka);               // ds_load from LDS slab
      }
      acc[ms] = __builtin_amdgcn_wmma_f32_16x16x32_bf16(false, a.v, false, b.v,
                                                        (short)0, acc[ms], false, false);
    }
  }
  if (tn < tiles_n) {
#pragma unroll
    for (int ms = 0; ms < 5; ++ms) {
      size_t cbase = (size_t)(row0 + ms * 16) * Np + (size_t)tn * 16 + l15;
#pragma unroll
      for (int j = 0; j < 8; ++j) {
        int mm = j + lh * 8;
        C[cbase + (size_t)mm * Np] = acc[ms][j];
      }
    }
  }
}

// ---------------- elementwise / graph kernels ------------------------------

__global__ void k_fill(float* __restrict__ p, float v, long long n) {
  long long t = (long long)blockIdx.x * blockDim.x + threadIdx.x;
  if (t < n) p[t] = v;
}

__global__ void k_deg_init(float* __restrict__ deg, int n) {
  int t = blockIdx.x * blockDim.x + threadIdx.x;
  if (t < n) deg[t] = 1.0f;                       // self loop
}

__global__ void k_deg_acc(const int* __restrict__ col, float* __restrict__ deg, int E) {
  int t = blockIdx.x * blockDim.x + threadIdx.x;
  if (t < E) atomicAdd(&deg[col[t]], 1.0f);
}

__global__ void k_rsqrt_ip(float* __restrict__ d, int n) {
  int t = blockIdx.x * blockDim.x + threadIdx.x;
  if (t < n) d[t] = rsqrtf(d[t]);
}

__global__ void k_edge_norm(const int* __restrict__ row, const int* __restrict__ col,
                            const float* __restrict__ dinv, float* __restrict__ nrm, int E) {
  int t = blockIdx.x * blockDim.x + threadIdx.x;
  if (t < E) nrm[t] = dinv[row[t]] * dinv[col[t]];
}

// H[i][f] = XW[i][f] * dinv[i]^2   (self-loop term; zero-inits padded cols)
__global__ void k_selfloop(const float* __restrict__ XW, const float* __restrict__ dinv,
                           float* __restrict__ H, int N, int Fp) {
  long long t = (long long)blockIdx.x * blockDim.x + threadIdx.x;
  long long total = (long long)N * Fp;
  if (t >= total) return;
  int i = (int)(t / Fp);
  float d = dinv[i];
  H[t] = XW[t] * d * d;
}

// H[col[e]][f] += XW[row[e]][f] * norm[e]
__global__ void k_scatter(const int* __restrict__ row, const int* __restrict__ col,
                          const float* __restrict__ nrm, const float* __restrict__ XW,
                          float* __restrict__ H, int E, int F, int Fp) {
  long long t = (long long)blockIdx.x * blockDim.x + threadIdx.x;
  long long total = (long long)E * F;
  if (t >= total) return;
  int e = (int)(t / F);
  int f = (int)(t % F);
  float v = XW[(size_t)row[e] * Fp + f] * nrm[e];
  atomicAdd(&H[(size_t)col[e] * Fp + f], v);
}

__global__ void k_bias_act(float* __restrict__ H, const float* __restrict__ bias,
                           long long rows, int F, int Fp, int do_relu) {
  long long t = (long long)blockIdx.x * blockDim.x + threadIdx.x;
  long long total = rows * Fp;
  if (t >= total) return;
  int f = (int)(t % Fp);
  float v = H[t] + ((f < F) ? bias[f] : 0.0f);
  if (do_relu) v = fmaxf(v, 0.0f);
  H[t] = v;
}

// segment max over non-negative values via monotone u32 bit pattern
__global__ void k_segmax(const float* __restrict__ H, const int* __restrict__ bvec,
                         float* __restrict__ g, int N, int Fp) {
  long long t = (long long)blockIdx.x * blockDim.x + threadIdx.x;
  long long total = (long long)N * Fp;
  if (t >= total) return;
  int i = (int)(t / Fp);
  int f = (int)(t % Fp);
  float v = fmaxf(H[t], 0.0f);
  atomicMax((unsigned int*)&g[(size_t)bvec[i] * Fp + f], __float_as_uint(v));
}

// fused conv1d (VALID, stride) + maxpool(Pk); layouts: in (B,Cin,Lin), w (Cout,Cin,K)
__global__ void k_convpool(const float* __restrict__ in, const float* __restrict__ w,
                           const float* __restrict__ bias, float* __restrict__ out,
                           int Cin, int Lin, int Cout, int K, int stride, int Pk, int Lout) {
  long long t = (long long)blockIdx.x * blockDim.x + threadIdx.x;
  long long total = (long long)256 * Cout * Lout;
  if (t >= total) return;
  int p  = (int)(t % Lout);
  int co = (int)((t / Lout) % Cout);
  int bb = (int)(t / ((long long)Lout * Cout));
  const float* wb = w + (size_t)co * Cin * K;
  const float* ib = in + (size_t)bb * Cin * Lin;
  float bv = bias[co];
  float best = -3.402823466e+38f;
  for (int pk = 0; pk < Pk; ++pk) {
    int q0 = (p * Pk + pk) * stride;
    float acc = bv;
    for (int ci = 0; ci < Cin; ++ci) {
      const float* ip = ib + (size_t)ci * Lin + q0;
      const float* wp = wb + (size_t)ci * K;
      for (int k = 0; k < K; ++k) acc += ip[k] * wp[k];
    }
    best = fmaxf(best, acc);
  }
  out[((size_t)bb * Cout + co) * Lout + p] = best;
}

__global__ void k_concat3(const float* __restrict__ a, const float* __restrict__ b,
                          const float* __restrict__ c, float* __restrict__ out, int rows, int w) {
  long long t = (long long)blockIdx.x * blockDim.x + threadIdx.x;
  long long total = (long long)rows * 3 * w;
  if (t >= total) return;
  int cc = (int)(t % (3 * w));
  long long r = t / (3 * w);
  float v;
  if (cc < w)           v = a[r * w + cc];
  else if (cc < 2 * w)  v = b[r * w + (cc - w)];
  else                  v = c[r * w + (cc - 2 * w)];
  out[t] = v;
}

__global__ void k_final(const float* __restrict__ f2, const float* __restrict__ W,
                        const float* __restrict__ bias, float* __restrict__ out, int K) {
  int r = blockIdx.x * blockDim.x + threadIdx.x;
  if (r >= 256) return;
  float acc = bias[0];
  for (int k = 0; k < K; ++k) acc += f2[(size_t)r * K + k] * W[k];
  out[r] = acc;
}

// ---------------------------------------------------------------------------

extern "C" void kernel_launch(void* const* d_in, const int* in_sizes, int n_in,
                              void* d_out, int out_size, void* d_ws, size_t ws_size,
                              hipStream_t stream) {
  const int N = 50000, E = 800000, Bq = 256;

  const float* x1   = (const float*)d_in[0];
  const int*   ei1  = (const int*)d_in[1];
  const int*   bv1  = (const int*)d_in[2];
  const float* x2   = (const float*)d_in[3];
  const int*   ei2  = (const int*)d_in[4];
  const int*   bv2  = (const int*)d_in[5];
  const float* tgt  = (const float*)d_in[6];
  const float* W1p  = (const float*)d_in[7];   const float* b1p  = (const float*)d_in[8];
  const float* W2p  = (const float*)d_in[9];   const float* b2p  = (const float*)d_in[10];
  const float* W3p  = (const float*)d_in[11];  const float* b3p  = (const float*)d_in[12];
  const float* Wg1p = (const float*)d_in[13];  const float* bg1p = (const float*)d_in[14];
  const float* Wg2p = (const float*)d_in[15];  const float* bg2p = (const float*)d_in[16];
  const float* cw1p = (const float*)d_in[17];  const float* cb1p = (const float*)d_in[18];
  const float* cw2p = (const float*)d_in[19];  const float* cb2p = (const float*)d_in[20];
  const float* cw3p = (const float*)d_in[21];  const float* cb3p = (const float*)d_in[22];
  const float* cw4p = (const float*)d_in[23];  const float* cb4p = (const float*)d_in[24];
  const float* cw5p = (const float*)d_in[25];  const float* cb5p = (const float*)d_in[26];
  const float* cw6p = (const float*)d_in[27];  const float* cb6p = (const float*)d_in[28];
  const float* cw7p = (const float*)d_in[29];  const float* cb7p = (const float*)d_in[30];
  const float* Wxtp = (const float*)d_in[31];  const float* bxtp = (const float*)d_in[32];
  const float* Wf1p = (const float*)d_in[33];  const float* bf1p = (const float*)d_in[34];
  const float* Wf2p = (const float*)d_in[35];  const float* bf2p = (const float*)d_in[36];
  const float* Woutp= (const float*)d_in[37];  const float* boutp= (const float*)d_in[38];
  float* out = (float*)d_out;

  // ---- workspace carve (bump allocator, 256B aligned) ----
  char* base = (char*)d_ws;
  size_t off = 0;
  auto carve = [&](size_t bytes) -> void* {
    off = (off + 255) & ~(size_t)255;
    void* p = base + off;
    off += bytes;
    return p;
  };
  uint16_t* W1t  = (uint16_t*)carve((size_t)96   * 96   * 2);
  uint16_t* W2t  = (uint16_t*)carve((size_t)160  * 96   * 2);
  uint16_t* W3t  = (uint16_t*)carve((size_t)320  * 160  * 2);
  uint16_t* Wg1t = (uint16_t*)carve((size_t)1024 * 320  * 2);
  uint16_t* Wg2t = (uint16_t*)carve((size_t)128  * 1024 * 2);
  uint16_t* Wxtt = (uint16_t*)carve((size_t)128  * 512  * 2);
  uint16_t* Wf1t = (uint16_t*)carve((size_t)1024 * 384  * 2);
  uint16_t* Wf2t = (uint16_t*)carve((size_t)128  * 1024 * 2);
  float* dinv  = (float*)carve((size_t)N * 4);
  float* enorm = (float*)carve((size_t)E * 4);
  // region A: Hbf (N x 320 bf16) + XW (N x 320 f32); reused as conv ping buffer
  size_t hbfBytes = (((size_t)N * 320 * 2) + 255) & ~(size_t)255;
  char* regionA = (char*)carve(hbfBytes + (size_t)N * 320 * 4);
  uint16_t* Hbf  = (uint16_t*)regionA;
  float*    XW   = (float*)(regionA + hbfBytes);
  float*    convA = (float*)regionA;
  // region B: HA (N x 320 f32); reused as conv pong buffer
  char* regionB = (char*)carve((size_t)N * 320 * 4);
  float* HA    = (float*)regionB;
  float* convB = (float*)regionB;
  float* HB    = (float*)carve((size_t)N * 320 * 4);
  float*    gmax   = (float*)carve((size_t)Bq * 320 * 4);
  uint16_t* gbf    = (uint16_t*)carve((size_t)Bq * 320 * 2);
  float*    gt1    = (float*)carve((size_t)Bq * 1024 * 4);
  uint16_t* gt1bf  = (uint16_t*)carve((size_t)Bq * 1024 * 2);
  float*    g1     = (float*)carve((size_t)Bq * 128 * 4);
  float*    g2     = (float*)carve((size_t)Bq * 128 * 4);
  float*    xt     = (float*)carve((size_t)Bq * 128 * 4);
  uint16_t* xinbf  = (uint16_t*)carve((size_t)Bq * 512 * 2);
  float*    xcat   = (float*)carve((size_t)Bq * 384 * 4);
  uint16_t* xcatbf = (uint16_t*)carve((size_t)Bq * 384 * 2);
  float*    f1     = (float*)carve((size_t)Bq * 1024 * 4);
  uint16_t* f1bf   = (uint16_t*)carve((size_t)Bq * 1024 * 2);
  float*    f2     = (float*)carve((size_t)Bq * 128 * 4);

  auto g1d = [&](long long cnt) { return dim3((unsigned)((cnt + 255) / 256)); };

  auto cvt_pad = [&](const float* in, uint16_t* o, int rows, int cols, int cols_pad) {
    long long cnt = (long long)rows * cols_pad;
    k_cvt_pad<<<g1d(cnt), 256, 0, stream>>>(in, o, rows, cols, cols_pad);
  };
  auto cvt_wt = [&](const float* w, uint16_t* o, int K, int Nn, int Kp, int Np) {
    long long cnt = (long long)Np * Kp;
    k_cvt_wt<<<g1d(cnt), 256, 0, stream>>>(w, o, K, Nn, Kp, Np);
  };
  auto gemm = [&](const uint16_t* A, const uint16_t* Bt, float* C, int M, int Kp, int Np) {
    int tn = Np / 16;
    if ((M % TDM_BM) == 0 && Kp <= 320) {
      dim3 g((unsigned)(M / TDM_BM), (unsigned)((tn + 7) / 8));
      k_gemm_tdm<<<g, 256, 0, stream>>>(A, Bt, C, Kp, Np, tn);
    } else {
      int tm = M / 16;
      long long waves = (long long)tm * tn;
      k_gemm_bf16<<<(unsigned)((waves + 7) / 8), 256, 0, stream>>>(A, Bt, C, tm, tn, Kp, Np);
    }
  };
  auto bias_act = [&](float* H, const float* b, long long rows, int F, int Fp, int relu) {
    long long cnt = rows * Fp;
    k_bias_act<<<g1d(cnt), 256, 0, stream>>>(H, b, rows, F, Fp, relu);
  };

  // ---- one-time weight conversions (transposed, padded) ----
  cvt_wt(W1p,  W1t,  78,   78,   96,   96);
  cvt_wt(W2p,  W2t,  78,   156,  96,   160);
  cvt_wt(W3p,  W3t,  156,  312,  160,  320);
  cvt_wt(Wg1p, Wg1t, 312,  1024, 320,  1024);
  cvt_wt(Wg2p, Wg2t, 1024, 128,  1024, 128);
  cvt_wt(Wxtp, Wxtt, 512,  128,  512,  128);
  cvt_wt(Wf1p, Wf1t, 384,  1024, 384,  1024);
  cvt_wt(Wf2p, Wf2t, 1024, 128,  1024, 128);

  // ---- GCN branch ----
  auto run_branch = [&](const float* X, const int* ei, const int* bvec, float* gout) {
    const int* rowp = ei;
    const int* colp = ei + E;
    k_deg_init<<<g1d(N), 256, 0, stream>>>(dinv, N);
    k_deg_acc<<<g1d(E), 256, 0, stream>>>(colp, dinv, E);
    k_rsqrt_ip<<<g1d(N), 256, 0, stream>>>(dinv, N);
    k_edge_norm<<<g1d(E), 256, 0, stream>>>(rowp, colp, dinv, enorm, E);

    auto layer = [&](const float* Hin, int Fin, int FinP, const uint16_t* Wt,
                     const float* bias, int Fout, int FoutP, float* Hout) {
      cvt_pad(Hin, Hbf, N, Fin, FinP);
      gemm(Hbf, Wt, XW, N, FinP, FoutP);
      long long cnt = (long long)N * FoutP;
      k_selfloop<<<g1d(cnt), 256, 0, stream>>>(XW, dinv, Hout, N, FoutP);
      long long scnt = (long long)E * Fout;
      k_scatter<<<g1d(scnt), 256, 0, stream>>>(rowp, colp, enorm, XW, Hout, E, Fout, FoutP);
      bias_act(Hout, bias, N, Fout, FoutP, 1);
    };
    layer(X,  78,  96,  W1t, b1p, 78,  96,  HA);
    layer(HA, 96,  96,  W2t, b2p, 156, 160, HB);
    layer(HB, 160, 160, W3t, b3p, 312, 320, HA);

    k_fill<<<g1d((long long)Bq * 320), 256, 0, stream>>>(gmax, 0.0f, (long long)Bq * 320);
    k_segmax<<<g1d((long long)N * 320), 256, 0, stream>>>(HA, bvec, gmax, N, 320);

    cvt_pad(gmax, gbf, Bq, 320, 320);
    gemm(gbf, Wg1t, gt1, Bq, 320, 1024);
    bias_act(gt1, bg1p, Bq, 1024, 1024, 1);
    cvt_pad(gt1, gt1bf, Bq, 1024, 1024);
    gemm(gt1bf, Wg2t, gout, Bq, 1024, 128);
    bias_act(gout, bg2p, Bq, 128, 128, 0);
  };
  run_branch(x1, ei1, bv1, g1);
  run_branch(x2, ei2, bv2, g2);

  // ---- conv tower (regions A/B are free now) ----
  struct ConvSt { const float* w; const float* bias; int Cin, Cout, K, stride, Pk; };
  ConvSt cs[7] = {
    {cw1p, cb1p, 1,   32,  8, 2, 3}, {cw2p, cb2p, 32,  32,  8, 2, 3},
    {cw3p, cb3p, 32,  64,  4, 1, 3}, {cw4p, cb4p, 64,  64,  4, 1, 2},
    {cw5p, cb5p, 64,  128, 4, 1, 2}, {cw6p, cb6p, 128, 128, 2, 1, 2},
    {cw7p, cb7p, 128, 64,  2, 1, 2}};
  const float* csrc = tgt;
  int Lin = 17000;
  for (int i = 0; i < 7; ++i) {
    int Lconv = (Lin - cs[i].K) / cs[i].stride + 1;
    int Lout  = Lconv / cs[i].Pk;
    float* dst = (i & 1) ? convB : convA;
    long long cnt = (long long)Bq * cs[i].Cout * Lout;
    k_convpool<<<g1d(cnt), 256, 0, stream>>>(csrc, cs[i].w, cs[i].bias, dst,
                                             cs[i].Cin, Lin, cs[i].Cout,
                                             cs[i].K, cs[i].stride, cs[i].Pk, Lout);
    csrc = dst;
    Lin = Lout;
  }
  // csrc = convA, shape (256, 64, 8) == (256, 512) row-major
  cvt_pad(csrc, xinbf, Bq, 512, 512);
  gemm(xinbf, Wxtt, xt, Bq, 512, 128);
  bias_act(xt, bxtp, Bq, 128, 128, 0);

  // ---- head ----
  k_concat3<<<g1d((long long)Bq * 384), 256, 0, stream>>>(g1, g2, xt, xcat, Bq, 128);
  cvt_pad(xcat, xcatbf, Bq, 384, 384);
  gemm(xcatbf, Wf1t, f1, Bq, 384, 1024);
  bias_act(f1, bf1p, Bq, 1024, 1024, 1);
  cvt_pad(f1, f1bf, Bq, 1024, 1024);
  gemm(f1bf, Wf2t, f2, Bq, 1024, 128);
  bias_act(f2, bf2p, Bq, 128, 128, 1);
  k_final<<<1, 256, 0, stream>>>(f2, Woutp, boutp, out, 128);
}
